// Coord_VAE_seq_26688926777404
// MI455X (gfx1250) — matrix operand order repaired and checked by reading, not compiled, not run on hardware
//
#include <hip/hip_runtime.h>
#include <hip/hip_bf16.h>

// ---------------------------------------------------------------------------
// CDNA5 (gfx1250) Coord-VAE-seq. Heavy GEMM work on v_wmma_f32_16x16x32_f16
// with guard-free float4 (b128) fragment loads; cheap dim-32 ops as fused
// wave32 VALU kernels.
// ---------------------------------------------------------------------------

typedef __attribute__((ext_vector_type(16))) _Float16 v16h;
typedef __attribute__((ext_vector_type(8)))  float    v8f;

#define LL 256
#define PP 65536

__device__ __forceinline__ float lrelu_f(float x) { return x >= 0.f ? x : 0.01f * x; }
__device__ __forceinline__ float gelu_f(float x) {
  float x3 = x * x * x;
  return 0.5f * x * (1.f + tanhf(0.7978845608028654f * (x + 0.044715f * x3)));
}
__device__ __forceinline__ void pack8(v16h& d, int base, float4 u, float4 v) {
  d[base + 0] = (_Float16)u.x; d[base + 1] = (_Float16)u.y;
  d[base + 2] = (_Float16)u.z; d[base + 3] = (_Float16)u.w;
  d[base + 4] = (_Float16)v.x; d[base + 5] = (_Float16)v.y;
  d[base + 6] = (_Float16)v.z; d[base + 7] = (_Float16)v.w;
}

// ---------------------------------------------------------------------------
// Fast WMMA GEMM: C[M,N] = act(A[M,K] @ W[N,K]^T + bias[N])
// Requires K%32==0, N%16==0, M%64==0. Unconditional b128 fragment loads.
// Block = 128 (4 waves, one 16-row sub-tile each); grid = (N/16, M/64).
// ---------------------------------------------------------------------------
__global__ void gemm_wmma_fast(const float* __restrict__ A, const float* __restrict__ W,
                               const float* __restrict__ bias, float* __restrict__ C,
                               int M, int N, int K, int act) {
  const int lane = threadIdx.x & 31;
  const int wv   = threadIdx.x >> 5;
  const int m0   = (blockIdx.y * 4 + wv) * 16;
  const int n0   = blockIdx.x * 16;
  const int g    = lane >> 4;
  const int hn   = lane & 15;
  const float* arow = A + (long)(m0 + hn) * K;
  const float* wrow = W + (long)(n0 + hn) * K;

  v8f acc = {0.f, 0.f, 0.f, 0.f, 0.f, 0.f, 0.f, 0.f};
  for (int k0 = 0; k0 < K; k0 += 32) {
    v16h a, b;
    // A 16x32 f16 layout: lane holds k = {g*8..g*8+7} and {16+g*8..16+g*8+7}
    const float4* pa0 = (const float4*)(arow + k0 + g * 8);
    const float4* pa1 = (const float4*)(arow + k0 + 16 + g * 8);
    pack8(a, 0, pa0[0], pa0[1]);
    pack8(a, 8, pa1[0], pa1[1]);
    // B 32x16 f16 layout: lane holds k = {g*8..g*8+7} and {16+g*8..16+g*8+7}
    const float4* pb0 = (const float4*)(wrow + k0 + g * 8);
    const float4* pb1 = (const float4*)(wrow + k0 + 16 + g * 8);
    pack8(b, 0, pb0[0], pb0[1]);
    pack8(b, 8, pb1[0], pb1[1]);
    acc = __builtin_amdgcn_wmma_f32_16x16x32_f16(false, a, false, b, (short)0, acc,
                                                 false, false);
  }
  const int n = n0 + hn;
  float bs = bias ? bias[n] : 0.f;
#pragma unroll
  for (int r = 0; r < 8; ++r) {   // D: lanes 0-15 -> M=r, lanes 16-31 -> M=8+r
    int m = m0 + g * 8 + r;
    float x = acc[r] + bs;
    if (act == 1) x = gelu_f(x);
    C[(long)m * N + n] = x;
  }
}

// Guarded WMMA GEMM for the tiny heads (N=12/20, K=20 cases). M%16==0.
__global__ void gemm_wmma_safe(const float* __restrict__ A, const float* __restrict__ W,
                               const float* __restrict__ bias, float* __restrict__ C,
                               int M, int N, int K, int act) {
  const int lane = threadIdx.x & 31;
  const int wv   = threadIdx.x >> 5;
  const int m0   = (blockIdx.y * 4 + wv) * 16;
  const int n0   = blockIdx.x * 16;
  if (m0 >= M) return;  // wave-uniform
  const int g  = lane >> 4;
  const int hn = lane & 15;
  v8f acc = {0.f, 0.f, 0.f, 0.f, 0.f, 0.f, 0.f, 0.f};
  for (int k0 = 0; k0 < K; k0 += 32) {
    v16h a, b;
    const int m = m0 + hn;
#pragma unroll
    for (int v = 0; v < 8; ++v) {
      int kb = (v < 4) ? (g * 8 + v * 2) : (16 + g * 8 + (v - 4) * 2);
      int k  = k0 + kb;
      float a0 = (k     < K) ? A[(long)m * K + k]     : 0.f;
      float a1 = (k + 1 < K) ? A[(long)m * K + k + 1] : 0.f;
      a[2 * v] = (_Float16)a0; a[2 * v + 1] = (_Float16)a1;
    }
    const int n = n0 + hn;
#pragma unroll
    for (int v = 0; v < 8; ++v) {
      int k = k0 + ((v < 4) ? (g * 8 + v * 2) : (16 + g * 8 + (v - 4) * 2));
      float b0 = (n < N && k     < K) ? W[(long)n * K + k]     : 0.f;
      float b1 = (n < N && k + 1 < K) ? W[(long)n * K + k + 1] : 0.f;
      b[2 * v] = (_Float16)b0; b[2 * v + 1] = (_Float16)b1;
    }
    acc = __builtin_amdgcn_wmma_f32_16x16x32_f16(false, a, false, b, (short)0, acc,
                                                 false, false);
  }
  const int n = n0 + hn;
  if (n < N) {
    float bs = bias ? bias[n] : 0.f;
#pragma unroll
    for (int r = 0; r < 8; ++r) {
      int m = m0 + g * 8 + r;
      float x = acc[r] + bs;
      if (act == 1) x = gelu_f(x);
      C[(long)m * N + n] = x;
    }
  }
}

// ---------------------------------------------------------------------------
// Implicit-GEMM 3x3 conv2d, channel-last activations [pos][IC], halo-padded
// buffers (border loads unconditional, zeroed via selects). Weights pre-
// transposed to Wp[tap][oc][ic]. IC,OC multiples of 32/16. Optional fused
// InstanceNorm+lrelu on input. grid=(1024, OC/16), block=128.
// ---------------------------------------------------------------------------
__device__ __forceinline__ float4 norm4(float4 v, float4 m, float4 r) {
  v.x = lrelu_f((v.x - m.x) * r.x); v.y = lrelu_f((v.y - m.y) * r.y);
  v.z = lrelu_f((v.z - m.z) * r.z); v.w = lrelu_f((v.w - m.w) * r.w);
  return v;
}
__device__ __forceinline__ float4 sel4(bool c, float4 v) {
  float4 z; z.x = c ? v.x : 0.f; z.y = c ? v.y : 0.f;
  z.z = c ? v.z : 0.f; z.w = c ? v.w : 0.f; return z;
}

__global__ void conv2d_wmma(const float* __restrict__ X, const float* __restrict__ Wp,
                            const float* __restrict__ bias, float* __restrict__ Y,
                            int IC, int OC,
                            const float* __restrict__ mean, const float* __restrict__ rstd) {
  const int lane = threadIdx.x & 31;
  const int wv   = threadIdx.x >> 5;
  const int pos0 = (blockIdx.x * 4 + wv) * 16;
  const int oc0  = blockIdx.y * 16;
  const int g    = lane >> 4;
  const int hn   = lane & 15;
  const int pos  = pos0 + hn, py = pos >> 8, px = pos & 255;

  v8f acc = {0.f, 0.f, 0.f, 0.f, 0.f, 0.f, 0.f, 0.f};
  for (int t = 0; t < 9; ++t) {
    const int dy = t / 3 - 1, dx = t % 3 - 1;
    const bool inb = ((unsigned)(py + dy) < 256u) && ((unsigned)(px + dx) < 256u);
    const float* xrow = X + (long)(pos + dy * LL + dx) * IC;   // halo-safe
    const float* wrow = Wp + (long)(t * OC + oc0 + hn) * IC;
    for (int k0 = 0; k0 < IC; k0 += 32) {
      v16h a, b;
      const float4* pa0 = (const float4*)(wrow + k0 + g * 16);
      pack8(a, 0, pa0[0], pa0[1]);
      pack8(a, 8, pa0[2], pa0[3]);

      const int kb = k0 + g * 16;
      const float4* pb = (const float4*)(xrow + kb);
      float4 b0 = pb[0], b1 = pb[1], b2 = pb[2], b3 = pb[3];
      if (mean) {  // uniform branch
        const float4* pm = (const float4*)(mean + kb);
        const float4* pr = (const float4*)(rstd + kb);
        b0 = norm4(b0, pm[0], pr[0]); b1 = norm4(b1, pm[1], pr[1]);
        b2 = norm4(b2, pm[2], pr[2]); b3 = norm4(b3, pm[3], pr[3]);
      }
      b0 = sel4(inb, b0); b1 = sel4(inb, b1); b2 = sel4(inb, b2); b3 = sel4(inb, b3);
      pack8(b, 0, b0, b1);
      pack8(b, 8, b2, b3);
      acc = __builtin_amdgcn_wmma_f32_16x16x32_f16(false, a, false, b, (short)0, acc,
                                                   false, false);
    }
  }
  const float4* pbias = (const float4*)(bias + oc0 + g * 8);
  float4 bl = pbias[0], bh = pbias[1];
  float4 lo, hi;
  lo.x = acc[0] + bl.x; lo.y = acc[1] + bl.y; lo.z = acc[2] + bl.z; lo.w = acc[3] + bl.w;
  hi.x = acc[4] + bh.x; hi.y = acc[5] + bh.y; hi.z = acc[6] + bh.z; hi.w = acc[7] + bh.w;
  float* yp = Y + (long)pos * OC + oc0 + g * 8;
  *(float4*)yp       = lo;
  *(float4*)(yp + 4) = hi;
}

// Weight transpose: Wp[t][oc][ic] (ic zero-padded ICs -> ICp). grid=(OC,9), block=ICp.
__global__ void wprep2d(const float* __restrict__ W, float* __restrict__ Wp,
                        int ICp, int ICs, int OC) {
  const int oc = blockIdx.x, t = blockIdx.y, ic = threadIdx.x;
  Wp[((long)t * OC + oc) * ICp + ic] = (ic < ICs) ? W[(oc * ICs + ic) * 9 + t] : 0.f;
}

// Stage edge_feat [i,j,19] -> chlast [pos][32] zero-padded. grid=(256,256), block=32.
__global__ void stage_edge(const float* __restrict__ E, float* __restrict__ S) {
  const int i = blockIdx.x, j = blockIdx.y, c = threadIdx.x;
  S[((long)i * LL + j) * 32 + c] = (c < 19) ? E[((long)i * LL + j) * 19 + c] : 0.f;
}

// Per-channel mean/rstd over 65536 positions, channel-last input.
__global__ void stats2d(const float* __restrict__ Y, int OC, float* __restrict__ mean,
                        float* __restrict__ rstd) {
  const int c = blockIdx.x, t = threadIdx.x;
  float s = 0.f, q = 0.f;
  for (int i = t; i < PP; i += 256) { float v = Y[(long)i * OC + c]; s += v; q += v * v; }
  __shared__ float ss[256], qq[256];
  ss[t] = s; qq[t] = q; __syncthreads();
  for (int o = 128; o > 0; o >>= 1) {
    if (t < o) { ss[t] += ss[t + o]; qq[t] += qq[t + o]; }
    __syncthreads();
  }
  if (t == 0) {
    float m = ss[0] * (1.f / PP);
    float v = qq[0] * (1.f / PP) - m * m;
    mean[c] = m; rstd[c] = rsqrtf(v + 1e-5f);
  }
}

// Fused conv1d(3,pad=1) + InstanceNorm + lrelu. One block per out channel.
__global__ void conv1d_in_lrelu(const float* __restrict__ X, const float* __restrict__ Wt,
                                const float* __restrict__ bias, float* __restrict__ Y,
                                int IC, int in_chlast) {
  const int oc = blockIdx.x, pos = threadIdx.x;
  float acc = bias[oc];
  for (int ic = 0; ic < IC; ++ic) {
#pragma unroll
    for (int t = 0; t < 3; ++t) {
      int p = pos + t - 1;
      if (p >= 0 && p < LL) {
        float xv = in_chlast ? X[p * IC + ic] : X[ic * LL + p];
        acc += Wt[(oc * IC + ic) * 3 + t] * xv;
      }
    }
  }
  __shared__ float ss[256], qq[256];
  __shared__ float mS, rS;
  ss[pos] = acc; qq[pos] = acc * acc; __syncthreads();
  for (int o = 128; o > 0; o >>= 1) {
    if (pos < o) { ss[pos] += ss[pos + o]; qq[pos] += qq[pos + o]; }
    __syncthreads();
  }
  if (pos == 0) {
    float m = ss[0] * (1.f / LL);
    float v = qq[0] * (1.f / LL) - m * m;
    mS = m; rS = rsqrtf(v + 1e-5f);
  }
  __syncthreads();
  Y[oc * LL + pos] = lrelu_f((acc - mS) * rS);
}

// lat[i,c] = mean_j lrelu(norm(F2[(i,j),c]))  (channel-last F2, IC=256)
__global__ void lat_mean(const float* __restrict__ F2, const float* __restrict__ mean,
                         const float* __restrict__ rstd, float* __restrict__ lat) {
  const int i = blockIdx.x, c = threadIdx.x;
  float m = mean[c], r = rstd[c], s = 0.f;
  for (int j = 0; j < LL; ++j) s += lrelu_f((F2[((long)i * LL + j) * 256 + c] - m) * r);
  lat[i * 512 + c] = s * (1.f / LL);
}

// lat[i,256+c] = x1[c,i]
__global__ void lat_x1(const float* __restrict__ X1, float* __restrict__ lat) {
  const int i = blockIdx.x, c = threadIdx.x;
  lat[i * 512 + 256 + c] = X1[c * LL + i];
}

__global__ void reparam(const float* __restrict__ eps, const float* __restrict__ mu,
                        const float* __restrict__ lv, float* __restrict__ z) {
  int i = blockIdx.x * blockDim.x + threadIdx.x;
  z[i] = eps[i] * expf(0.5f * lv[i]) + mu[i];
}

// F2[(i,j),c] = F1[i,c]*F1[j,c]   (channel-last, 256 ch). grid=(256,256), block=256.
__global__ void outer2d(const float* __restrict__ F1, float* __restrict__ F2) {
  const int i = blockIdx.x, j = blockIdx.y, c = threadIdx.x;
  F2[((long)i * LL + j) * 256 + c] = F1[i * 256 + c] * F1[j * 256 + c];
}

// out2d[(i,j),c] = n(F2[(j,i),c]) + n(F2[(i,j),c])
__global__ void finalize2d(const float* __restrict__ F2, const float* __restrict__ mean,
                           const float* __restrict__ rstd, float* __restrict__ out2d) {
  const int i = blockIdx.x, j = blockIdx.y, c = threadIdx.x;  // c < 32
  float m = mean[c], r = rstd[c];
  float a = lrelu_f((F2[((long)j * LL + i) * 32 + c] - m) * r);
  float b = lrelu_f((F2[((long)i * LL + j) * 32 + c] - m) * r);
  out2d[((long)i * LL + j) * 32 + c] = a + b;
}

// LayerNorm over dim 32, one wave per row.
__global__ void layernorm32(const float* __restrict__ X, const float* __restrict__ gg,
                            const float* __restrict__ bb, float* __restrict__ Y) {
  const int r = blockIdx.x, c = threadIdx.x;
  float v = X[r * 32 + c];
  float s = v;
  for (int o = 16; o > 0; o >>= 1) s += __shfl_xor(s, o, 32);
  float m = s * (1.f / 32.f);
  float d = v - m, q = d * d;
  for (int o = 16; o > 0; o >>= 1) q += __shfl_xor(q, o, 32);
  float rs = rsqrtf(q * (1.f / 32.f) + 1e-5f);
  Y[r * 32 + c] = d * rs * gg[c] + bb[c];
}

// Interleaved-pair RoPE, 8 heads x 64 dims.
__global__ void rope512(float* __restrict__ X, int rowstride) {
  const int pos = blockIdx.x, t = threadIdx.x;
  const int head = t >> 5, p = t & 31;
  float inv = powf(10000.f, -(float)(2 * p) / 64.f);
  float s, c;
  sincosf((float)pos * inv, &s, &c);
  float* q = X + pos * rowstride + head * 64 + 2 * p;
  float x0 = q[0], x1 = q[1];
  q[0] = x0 * c - x1 * s;
  q[1] = x1 * c + x0 * s;
}

// Fused edge-conditioned attention for one (head, i).
__global__ void attn_fused(const float* __restrict__ q, const float* __restrict__ kv,
                           const float* __restrict__ e, float* __restrict__ out) {
  const int h = blockIdx.x, i = blockIdx.y, t = threadIdx.x;
  __shared__ float qs[64], pr[256], red[256];
  if (t < 64) qs[t] = q[i * 512 + h * 64 + t];
  __syncthreads();
  const float4* kp = (const float4*)(kv + t * 1024 + h * 64);
  const float4* ep = (const float4*)(e + ((long)i * LL + t) * 512 + h * 64);
  float s = 0.f;
#pragma unroll 4
  for (int d4 = 0; d4 < 16; ++d4) {
    float4 kk = kp[d4], ee = ep[d4];
    s += qs[4 * d4 + 0] * (kk.x + ee.x) + qs[4 * d4 + 1] * (kk.y + ee.y) +
         qs[4 * d4 + 2] * (kk.z + ee.z) + qs[4 * d4 + 3] * (kk.w + ee.w);
  }
  s *= 0.125f;
  pr[t] = s; red[t] = s; __syncthreads();
  for (int o = 128; o > 0; o >>= 1) { if (t < o) red[t] = fmaxf(red[t], red[t + o]); __syncthreads(); }
  float mx = red[0]; __syncthreads();
  float pe = expf(pr[t] - mx);
  pr[t] = pe; red[t] = pe; __syncthreads();
  for (int o = 128; o > 0; o >>= 1) { if (t < o) red[t] += red[t + o]; __syncthreads(); }
  float inv = 1.f / red[0];
  __syncthreads();
  const int d = t & 63, ch = t >> 6;
  float acc = 0.f;
  for (int j = ch * 64; j < ch * 64 + 64; ++j) {
    float p = pr[j] * inv;
    acc += p * (kv[j * 1024 + 512 + h * 64 + d] + e[((long)i * LL + j) * 512 + h * 64 + d]);
  }
  red[t] = acc; __syncthreads();
  if (t < 64) out[i * 512 + h * 64 + t] = red[t] + red[64 + t] + red[128 + t] + red[192 + t];
}

// nodes = x*g + nodes*(1-g), g = sigmoid([x, nodes, x-nodes] . gw)
__global__ void gated_res(float* __restrict__ nodes, const float* __restrict__ xo,
                          const float* __restrict__ gw) {
  const int r = blockIdx.x, c = threadIdx.x;
  float o = xo[r * 32 + c], n = nodes[r * 32 + c];
  float s = o * gw[c] + n * gw[32 + c] + (o - n) * gw[64 + c];
  for (int off = 16; off > 0; off >>= 1) s += __shfl_xor(s, off, 32);
  float g = 1.f / (1.f + expf(-s));
  nodes[r * 32 + c] = o * g + n * (1.f - g);
}

// [C,256] channel-first -> [256,C]
__global__ void transpose_cf(const float* __restrict__ X, float* __restrict__ Y, int C) {
  const int pos = blockIdx.x, c = threadIdx.x;
  Y[pos * C + c] = X[c * LL + pos];
}

// ---------------------------------------------------------------------------
// d_in flattening (JAX pytree: top-level dict in setup order; `params` leaves
// with dict keys sorted alphabetically, lists/tuples in order).
// ---------------------------------------------------------------------------
enum {
  IN_SCALAR = 0, IN_EDGE = 1, IN_EPS = 2,
  P_COORD_W = 3,
  P_DEC1D = 4,    // (w,b) x3 : 4..9
  P_DEC2D = 10,   // (w,b) x3 : 10..15
  P_ENC1D = 16,   // (w,b) x4 : 16..23
  P_ENC2D = 24,   // (w,b) x4 : 24..31
  P_FC_MU_B = 32, P_FC_MU_W = 33, P_FC_VAR_B = 34, P_FC_VAR_W = 35,
  P_LAYER0 = 36,  // 18 leaves per layer x 6
  P_SEQ_B1 = 144, P_SEQ_B2 = 145, P_SEQ_LN_B = 146, P_SEQ_LN_G = 147,
  P_SEQ_W1 = 148, P_SEQ_W2 = 149, P_UP_B = 150, P_UP_W = 151,
};
enum { A_BE = 0, A_BKV = 1, A_BO = 2, A_BQ = 3, A_GW = 4, A_LNB = 5, A_LNG = 6,
       A_WE = 7, A_WKV = 8, A_WO = 9, A_WQ = 10,
       F_B1 = 11, F_B2 = 12, F_GW = 13, F_LNB = 14, F_LNG = 15, F_W1 = 16, F_W2 = 17 };

extern "C" void kernel_launch(void* const* d_in, const int* in_sizes, int n_in,
                              void* d_out, int out_size, void* d_ws, size_t ws_size,
                              hipStream_t stream) {
  auto IN = [&](int i) -> const float* { return (const float*)d_in[i]; };
  float* ws   = (float*)d_ws;
  float* dout = (float*)d_out;

  // ---- workspace layout (float offsets); conv buffers halo-padded ----
  const long HALO = 66048;                       // > 257*256 floats
  float* bufA  = ws + HALO;                      // 16,777,216 chlast conv ping
  float* bufB  = ws + HALO + 16777216 + HALO;    // 16,777,216 chlast conv pong
  float* ebuf  = ws + 0;                         // 33,554,432 (aliases conv area)
  long   off   = HALO + 16777216 + HALO + 16777216 + HALO;  // 33,752,576
  float* out2d = ws + off;            off += 2097152;
  float* wbuf  = ws + off;            off += 589824;   // 9*256*256 max
  float* qbuf  = ws + off;            off += 131072;
  float* kvbuf = ws + off;            off += 262144;
  float* aobuf = ws + off;            off += 131072;
  float* xln   = ws + off;            off += 8192;
  float* h1    = ws + off;            off += 32768;
  float* obuf  = ws + off;            off += 8192;
  float* nodes = ws + off;            off += 8192;
  float* lat   = ws + off;            off += 131072;
  float* zbuf  = ws + off;            off += 16384;
  float* x1a   = ws + off;            off += 65536;
  float* x1b   = ws + off;            off += 65536;
  float* f1p   = ws + off;            off += 65536;
  float* sq1   = ws + off;            off += 5120;
  float* meanb = ws + off;            off += 256;
  float* rstdb = ws + off;            off += 256;

  // ---- output: coords(3072) | seqs(5120) | mu(16384) | logvar(16384) ----
  float* o_coord = dout + 0;
  float* o_seq   = dout + 3072;
  float* o_mu    = dout + 8192;
  float* o_lv    = dout + 24576;

  auto gemm = [&](const float* A, const float* W, const float* bias, float* C,
                  int M, int N, int K, int act) {
    gemm_wmma_fast<<<dim3(N / 16, M / 64), dim3(128), 0, stream>>>(A, W, bias, C, M, N, K, act);
  };
  auto gemm_s = [&](const float* A, const float* W, const float* bias, float* C,
                    int M, int N, int K, int act) {
    gemm_wmma_safe<<<dim3((N + 15) / 16, (M + 63) / 64), dim3(128), 0, stream>>>(A, W, bias, C, M, N, K, act);
  };
  auto conv2d = [&](const float* X, const float* W, const float* b, float* Y,
                    int ICp, int ICs, int OC, const float* mn, const float* rs) {
    wprep2d<<<dim3(OC, 9), dim3(ICp), 0, stream>>>(W, wbuf, ICp, ICs, OC);
    conv2d_wmma<<<dim3(1024, OC / 16), dim3(128), 0, stream>>>(X, wbuf, b, Y, ICp, OC, mn, rs);
  };

  // ================= 1D encoder ==========================================
  conv1d_in_lrelu<<<32,  256, 0, stream>>>(IN(IN_SCALAR), IN(P_ENC1D + 0), IN(P_ENC1D + 1), x1a, 26, 1);
  conv1d_in_lrelu<<<64,  256, 0, stream>>>(x1a, IN(P_ENC1D + 2), IN(P_ENC1D + 3), x1b, 32, 0);
  conv1d_in_lrelu<<<128, 256, 0, stream>>>(x1b, IN(P_ENC1D + 4), IN(P_ENC1D + 5), x1a, 64, 0);
  conv1d_in_lrelu<<<256, 256, 0, stream>>>(x1a, IN(P_ENC1D + 6), IN(P_ENC1D + 7), x1b, 128, 0);

  // ================= 2D encoder ==========================================
  stage_edge<<<dim3(256, 256), 32, 0, stream>>>(IN(IN_EDGE), bufA);
  conv2d(bufA, IN(P_ENC2D + 0), IN(P_ENC2D + 1), bufB, 32, 19, 32, nullptr, nullptr);
  stats2d<<<32, 256, 0, stream>>>(bufB, 32, meanb, rstdb);
  conv2d(bufB, IN(P_ENC2D + 2), IN(P_ENC2D + 3), bufA, 32, 32, 64, meanb, rstdb);
  stats2d<<<64, 256, 0, stream>>>(bufA, 64, meanb, rstdb);
  conv2d(bufA, IN(P_ENC2D + 4), IN(P_ENC2D + 5), bufB, 64, 64, 128, meanb, rstdb);
  stats2d<<<128, 256, 0, stream>>>(bufB, 128, meanb, rstdb);
  conv2d(bufB, IN(P_ENC2D + 6), IN(P_ENC2D + 7), bufA, 128, 128, 256, meanb, rstdb);
  stats2d<<<256, 256, 0, stream>>>(bufA, 256, meanb, rstdb);

  // ================= latent ==============================================
  lat_mean<<<256, 256, 0, stream>>>(bufA, meanb, rstdb, lat);
  lat_x1<<<256, 256, 0, stream>>>(x1b, lat);
  gemm(lat, IN(P_FC_MU_W),  IN(P_FC_MU_B),  o_mu, 256, 64, 512, 0);
  gemm(lat, IN(P_FC_VAR_W), IN(P_FC_VAR_B), o_lv, 256, 64, 512, 0);
  reparam<<<64, 256, 0, stream>>>(IN(IN_EPS), o_mu, o_lv, zbuf);
  gemm(zbuf, IN(P_UP_W), IN(P_UP_B), f1p, 256, 256, 64, 0);

  // ================= 2D decoder ==========================================
  outer2d<<<dim3(256, 256), 256, 0, stream>>>(f1p, bufA);
  conv2d(bufA, IN(P_DEC2D + 0), IN(P_DEC2D + 1), bufB, 256, 256, 128, nullptr, nullptr);
  stats2d<<<128, 256, 0, stream>>>(bufB, 128, meanb, rstdb);
  conv2d(bufB, IN(P_DEC2D + 2), IN(P_DEC2D + 3), bufA, 128, 128, 64, meanb, rstdb);
  stats2d<<<64, 256, 0, stream>>>(bufA, 64, meanb, rstdb);
  conv2d(bufA, IN(P_DEC2D + 4), IN(P_DEC2D + 5), bufB, 64, 64, 32, meanb, rstdb);
  stats2d<<<32, 256, 0, stream>>>(bufB, 32, meanb, rstdb);
  finalize2d<<<dim3(256, 256), 32, 0, stream>>>(bufB, meanb, rstdb, out2d);

  // ================= 1D decoder -> nodes =================================
  conv1d_in_lrelu<<<128, 256, 0, stream>>>(f1p, IN(P_DEC1D + 0), IN(P_DEC1D + 1), x1a, 256, 1);
  conv1d_in_lrelu<<<64,  256, 0, stream>>>(x1a, IN(P_DEC1D + 2), IN(P_DEC1D + 3), x1b, 128, 0);
  conv1d_in_lrelu<<<32,  256, 0, stream>>>(x1b, IN(P_DEC1D + 4), IN(P_DEC1D + 5), x1a, 64, 0);
  transpose_cf<<<256, 32, 0, stream>>>(x1a, nodes, 32);

  // ================= transformer (6 layers) ==============================
  for (int l = 0; l < 6; ++l) {
    const int B = P_LAYER0 + l * 18;
    layernorm32<<<256, 32, 0, stream>>>(nodes, IN(B + A_LNG), IN(B + A_LNB), xln);
    gemm(xln,   IN(B + A_WQ),  IN(B + A_BQ),  qbuf,  256,   512, 32, 0);
    gemm(xln,   IN(B + A_WKV), IN(B + A_BKV), kvbuf, 256,  1024, 32, 0);
    gemm(out2d, IN(B + A_WE),  IN(B + A_BE),  ebuf, 65536,  512, 32, 0);
    rope512<<<256, 256, 0, stream>>>(qbuf, 512);
    rope512<<<256, 256, 0, stream>>>(kvbuf, 1024);
    attn_fused<<<dim3(8, 256), 256, 0, stream>>>(qbuf, kvbuf, ebuf, aobuf);
    gemm(aobuf, IN(B + A_WO), IN(B + A_BO), obuf, 256, 32, 512, 0);
    gated_res<<<256, 32, 0, stream>>>(nodes, obuf, IN(B + A_GW));
    layernorm32<<<256, 32, 0, stream>>>(nodes, IN(B + F_LNG), IN(B + F_LNB), xln);
    gemm(xln, IN(B + F_W1), IN(B + F_B1), h1,   256, 128, 32, 1);
    gemm(h1,  IN(B + F_W2), IN(B + F_B2), obuf, 256, 32, 128, 0);
    gated_res<<<256, 32, 0, stream>>>(nodes, obuf, IN(B + F_GW));
  }

  // ================= heads ===============================================
  gemm_s(nodes, IN(P_COORD_W), nullptr, o_coord, 256, 12, 32, 0);
  layernorm32<<<256, 32, 0, stream>>>(nodes, IN(P_SEQ_LN_G), IN(P_SEQ_LN_B), xln);
  gemm_s(xln, IN(P_SEQ_W1), IN(P_SEQ_B1), sq1,   256, 20, 32, 1);
  gemm_s(sq1, IN(P_SEQ_W2), IN(P_SEQ_B2), o_seq, 256, 20, 20, 0);

  (void)in_sizes; (void)n_in; (void)out_size; (void)ws_size; (void)off;
}
